// Attention_25675314495536
// MI455X (gfx1250) — matrix operand order repaired and checked by reading, not compiled, not run on hardware
//
#include <hip/hip_runtime.h>
#include <hip/hip_fp16.h>

typedef __attribute__((ext_vector_type(16))) _Float16 v16h;
typedef __attribute__((ext_vector_type(8)))  float    v8f;
typedef int v4i_vec __attribute__((vector_size(16)));   /* matches builtin param type */

#define EMBED 768
#define NHEAD 12
#define HDIM  64
#define SEQ   4096
#define BATCH 2
#define M_TOTAL (BATCH * SEQ)   /* 8192 */
#define ATTN_SCALE 0.125f       /* 1/sqrt(64) */
#define NEG_BIG (-1.0e30f)
#define KSTRIDE 72              /* padded LDS row stride (halves): 144B, 16B aligned, conflict-free */

// gfx1250 async global->LDS path (ASYNCcnt-tracked), with sync fallback
#if defined(__gfx1250__) && defined(__has_builtin)
#if __has_builtin(__builtin_amdgcn_global_load_async_to_lds_b128) && \
    __has_builtin(__builtin_amdgcn_s_wait_asynccnt)
#define USE_ASYNC 1
#endif
#endif
#ifndef USE_ASYNC
#define USE_ASYNC 0
#endif

#define AS1 __attribute__((address_space(1)))
#define AS3 __attribute__((address_space(3)))

__device__ __forceinline__ void copy16_async(const _Float16* src, _Float16* lds_dst) {
#if USE_ASYNC
  __builtin_amdgcn_global_load_async_to_lds_b128(
      (AS1 v4i_vec*)(uintptr_t)src,
      (AS3 v4i_vec*)(uint32_t)(uintptr_t)lds_dst,  /* hw truncates flat LDS addr to [31:0] */
      /*imm offset=*/0, /*cpol=*/0);
#else
  *(uint4*)lds_dst = *(const uint4*)src;
#endif
}

__device__ __forceinline__ v8f wmma_f16(v16h a, v16h b, v8f c) {
  return __builtin_amdgcn_wmma_f32_16x16x32_f16(
      /*neg_a=*/false, a, /*neg_b=*/false, b,
      /*c_mod=*/(short)0, c, /*reuse_a=*/false, /*reuse_b=*/false);
}

// A-matrix (16x32 f16) element->K column for lane group g, element i
__device__ __forceinline__ int a_col(int i, int g) { return i + (i & 8) + 8 * g; }
// B-matrix (32x16 f16) element->K row for lane group g, element i
__device__ __forceinline__ int b_row(int i, int g) { return i + (i & 8) + 8 * g; }

// ---------------------------------------------------------------------------
// Kernel 1: qkv = x @ W_qkv^T  (f32 in, f16 out scattered to [B,H,S,D])
// Block: 128 threads (4 waves). Tile: 64 rows x 64 cols. K stepped by 32.
// ---------------------------------------------------------------------------
__global__ __launch_bounds__(128) void qkv_gemm(
    const float* __restrict__ x, const float* __restrict__ Wqkv,
    _Float16* __restrict__ Qw, _Float16* __restrict__ Kw,
    _Float16* __restrict__ Vw) {
  __shared__ __align__(16) _Float16 Wh[64][40];

  const int m0 = blockIdx.x * 64;
  const int n0 = blockIdx.y * 64;
  const int t = threadIdx.x;
  const int wave = t >> 5, lane = t & 31;
  const int g = lane >> 4, lm = lane & 15;

  v8f acc[4] = {{}, {}, {}, {}};
  const float* xrow = x + (size_t)(m0 + wave * 16 + lm) * EMBED;

  for (int k0 = 0; k0 < EMBED; k0 += 32) {
    {
      const int subrow = t >> 3;        // 0..15
      const int c4 = (t & 7) * 4;       // 0..28
#pragma unroll
      for (int p = 0; p < 4; ++p) {
        const int row = subrow + 16 * p;
        const float4 v = *(const float4*)(Wqkv + (size_t)(n0 + row) * EMBED + k0 + c4);
        Wh[row][c4 + 0] = (_Float16)v.x;
        Wh[row][c4 + 1] = (_Float16)v.y;
        Wh[row][c4 + 2] = (_Float16)v.z;
        Wh[row][c4 + 3] = (_Float16)v.w;
      }
    }
    __syncthreads();

    v16h a;
#pragma unroll
    for (int i = 0; i < 16; ++i) a[i] = (_Float16)xrow[k0 + a_col(i, g)];

#pragma unroll
    for (int nt = 0; nt < 4; ++nt) {
      v16h bf;
#pragma unroll
      for (int i = 0; i < 16; ++i) bf[i] = Wh[nt * 16 + lm][b_row(i, g)];
      acc[nt] = wmma_f16(a, bf, acc[nt]);
    }
    __syncthreads();
  }

  // Scatter into Q/K/V [B,H,S,D] f16 (fuses head split + transpose)
#pragma unroll
  for (int nt = 0; nt < 4; ++nt) {
#pragma unroll
    for (int r = 0; r < 8; ++r) {
      const int m = m0 + wave * 16 + r + 8 * g;
      const int f = n0 + nt * 16 + lm;
      const int bb = m >> 12;        // / SEQ
      const int s = m & (SEQ - 1);
      const int part = f / EMBED;
      const int fe = f % EMBED;
      const int h = fe >> 6, d = fe & 63;
      _Float16* dst = (part == 0) ? Qw : (part == 1) ? Kw : Vw;
      dst[((((size_t)bb * NHEAD) + h) * SEQ + s) * HDIM + d] = (_Float16)acc[nt][r];
    }
  }
}

// ---------------------------------------------------------------------------
// Kernel 2: flash attention. One WG = 4 waves = 64 query rows for one (b,h).
// Double-buffered K tiles via async global->LDS; V prefetched through VGPRs
// and stored transposed+padded so P*V B-fragments are contiguous b128 reads.
// ---------------------------------------------------------------------------
__global__ __launch_bounds__(128) void flash_attn(
    const _Float16* __restrict__ Qw, const _Float16* __restrict__ Kw,
    const _Float16* __restrict__ Vw, _Float16* __restrict__ Ao) {
  __shared__ __align__(16) _Float16 Kt[2][64 * KSTRIDE];  // [key][d], padded
  __shared__ __align__(16) _Float16 Vt[2][64 * KSTRIDE];  // transposed [d][key], padded
  __shared__ __align__(16) _Float16 Pb[4][16 * KSTRIDE];  // per-wave P tile

  const int idx = blockIdx.x;
  const int qb = idx & 63;               // SEQ/64 query blocks
  const int h = (idx >> 6) % NHEAD;
  const int bb = idx / (64 * NHEAD);
  const int t = threadIdx.x;
  const int wave = t >> 5, lane = t & 31;
  const int g = lane >> 4, lm = lane & 15;
  const int subrow = t >> 3;             // 0..15
  const int c8 = (t & 7) * 8;            // 0..56 halves

  const size_t head_off = (((size_t)bb * NHEAD) + h) * SEQ * HDIM;
  const _Float16* Qp = Qw + head_off;
  const _Float16* Kp = Kw + head_off;
  const _Float16* Vp = Vw + head_off;

  const int q0 = qb * 64 + wave * 16;

  // Q fragments: 16 rows x 64 dims = two 16x32 A-frags
  v16h qa[2];
#pragma unroll
  for (int f2 = 0; f2 < 2; ++f2)
#pragma unroll
    for (int i = 0; i < 16; ++i)
      qa[f2][i] = Qp[(size_t)(q0 + lm) * HDIM + a_col(i, g) + 32 * f2];

  v8f o[4] = {{}, {}, {}, {}};
  float mrow[8], lrow[8];
#pragma unroll
  for (int r = 0; r < 8; ++r) { mrow[r] = NEG_BIG; lrow[r] = 0.0f; }

  uint4 vreg[4];

  // ---- prologue: tile 0 into buffer 0 ----
#pragma unroll
  for (int p = 0; p < 4; ++p) {
    const int row = subrow + 16 * p;
    copy16_async(&Kp[(size_t)row * HDIM + c8], &Kt[0][row * KSTRIDE + c8]);
    vreg[p] = *(const uint4*)&Vp[(size_t)row * HDIM + c8];
  }
#pragma unroll
  for (int p = 0; p < 4; ++p) {
    const int row = subrow + 16 * p;            // key index
    const _Float16* hh = (const _Float16*)&vreg[p];
#pragma unroll
    for (int j = 0; j < 8; ++j) Vt[0][(c8 + j) * KSTRIDE + row] = hh[j];
  }

  const int NT = SEQ / 64;
  for (int it = 0; it < NT; ++it) {
    const int cur = it & 1;
    const bool pre = (it + 1 < NT);

    // issue next tile: K async -> LDS, V -> VGPRs
    if (pre) {
      const int kt0n = (it + 1) * 64;
#pragma unroll
      for (int p = 0; p < 4; ++p) {
        const int row = subrow + 16 * p;
        copy16_async(&Kp[(size_t)(kt0n + row) * HDIM + c8],
                     &Kt[cur ^ 1][row * KSTRIDE + c8]);
        vreg[p] = *(const uint4*)&Vp[(size_t)(kt0n + row) * HDIM + c8];
      }
    }
#if USE_ASYNC
    if (pre) __builtin_amdgcn_s_wait_asynccnt(4);   // current tile's 4 done (in-order)
    else     __builtin_amdgcn_s_wait_asynccnt(0);
#endif
    __syncthreads();   // Kt[cur] + Vt[cur] visible to all waves

    // Scores: 4 fragments of 16 keys, K-dim = 64 (two WMMAs each)
    v8f sf[4];
#pragma unroll
    for (int kt = 0; kt < 4; ++kt) {
      v16h b0, b1;
#pragma unroll
      for (int i = 0; i < 16; ++i) {
        const int d = b_row(i, g);
        b0[i] = Kt[cur][(kt * 16 + lm) * KSTRIDE + d];
        b1[i] = Kt[cur][(kt * 16 + lm) * KSTRIDE + d + 32];
      }
      v8f c = {};
      c = wmma_f16(qa[0], b0, c);
      c = wmma_f16(qa[1], b1, c);
      sf[kt] = c;
    }

    // Online softmax update (per lane: 8 rows x 4 key-frag columns)
    float alpha[8];
#pragma unroll
    for (int r = 0; r < 8; ++r) {
      float mx = sf[0][r] * ATTN_SCALE;
#pragma unroll
      for (int kt = 1; kt < 4; ++kt) mx = fmaxf(mx, sf[kt][r] * ATTN_SCALE);
#pragma unroll
      for (int off = 1; off < 16; off <<= 1)
        mx = fmaxf(mx, __shfl_xor(mx, off, 32));   // reduce within 16-lane half
      const float mnew = fmaxf(mrow[r], mx);
      alpha[r] = __expf(mrow[r] - mnew);
      float rs = 0.0f;
#pragma unroll
      for (int kt = 0; kt < 4; ++kt) {
        const float p = __expf(sf[kt][r] * ATTN_SCALE - mnew);
        sf[kt][r] = p;
        rs += p;
      }
#pragma unroll
      for (int off = 1; off < 16; off <<= 1) rs += __shfl_xor(rs, off, 32);
      lrow[r] = lrow[r] * alpha[r] + rs;
      mrow[r] = mnew;
    }

    // Write P (C-layout) to per-wave LDS buffer; rescale O accumulators
#pragma unroll
    for (int kt = 0; kt < 4; ++kt)
#pragma unroll
      for (int r = 0; r < 8; ++r)
        Pb[wave][(r + 8 * g) * KSTRIDE + kt * 16 + lm] = (_Float16)sf[kt][r];
#pragma unroll
    for (int dt = 0; dt < 4; ++dt)
#pragma unroll
      for (int r = 0; r < 8; ++r) o[dt][r] *= alpha[r];
    __syncthreads();

    // Re-read P as A-frags (layout conversion via LDS)
    v16h pa[2];
#pragma unroll
    for (int f2 = 0; f2 < 2; ++f2)
#pragma unroll
      for (int i = 0; i < 16; ++i)
        pa[f2][i] = Pb[wave][lm * KSTRIDE + a_col(i, g) + 32 * f2];

    // O += P * V  (V B-frags now contiguous thanks to transposed store)
#pragma unroll
    for (int dt = 0; dt < 4; ++dt) {
      v16h vb0, vb1;
#pragma unroll
      for (int i = 0; i < 16; ++i) {
        const int k = b_row(i, g);
        vb0[i] = Vt[cur][(dt * 16 + lm) * KSTRIDE + k];
        vb1[i] = Vt[cur][(dt * 16 + lm) * KSTRIDE + k + 32];
      }
      o[dt] = wmma_f16(pa[0], vb0, o[dt]);
      o[dt] = wmma_f16(pa[1], vb1, o[dt]);
    }
    __syncthreads();   // all reads of tile `cur` done

    // store prefetched V (transposed) into the other buffer
    if (pre) {
#pragma unroll
      for (int p = 0; p < 4; ++p) {
        const int row = subrow + 16 * p;
        const _Float16* hh = (const _Float16*)&vreg[p];
#pragma unroll
        for (int j = 0; j < 8; ++j)
          Vt[cur ^ 1][(c8 + j) * KSTRIDE + row] = hh[j];
      }
    }
  }

  // Normalize and store to attn-out [B,S,E] f16 (e = h*64 + d)
#pragma unroll
  for (int r = 0; r < 8; ++r) {
    const float inv = 1.0f / lrow[r];
    const int srow = q0 + r + 8 * g;
#pragma unroll
    for (int dt = 0; dt < 4; ++dt)
      Ao[((size_t)(bb * SEQ + srow)) * EMBED + h * HDIM + dt * 16 + lm] =
          (_Float16)(o[dt][r] * inv);
  }
}

// ---------------------------------------------------------------------------
// Kernel 3: out = attn_out @ W_proj^T + b_proj  (f16 A, f32 W/bias, f32 out)
// ---------------------------------------------------------------------------
__global__ __launch_bounds__(128) void proj_gemm(
    const _Float16* __restrict__ A, const float* __restrict__ Wp,
    const float* __restrict__ bias, float* __restrict__ out) {
  __shared__ __align__(16) _Float16 Wh[64][40];

  const int m0 = blockIdx.x * 64;
  const int n0 = blockIdx.y * 64;
  const int t = threadIdx.x;
  const int wave = t >> 5, lane = t & 31;
  const int g = lane >> 4, lm = lane & 15;

  v8f acc[4] = {{}, {}, {}, {}};
  const _Float16* arow = A + (size_t)(m0 + wave * 16 + lm) * EMBED;

  for (int k0 = 0; k0 < EMBED; k0 += 32) {
    {
      const int subrow = t >> 3;
      const int c4 = (t & 7) * 4;
#pragma unroll
      for (int p = 0; p < 4; ++p) {
        const int row = subrow + 16 * p;
        const float4 v = *(const float4*)(Wp + (size_t)(n0 + row) * EMBED + k0 + c4);
        Wh[row][c4 + 0] = (_Float16)v.x;
        Wh[row][c4 + 1] = (_Float16)v.y;
        Wh[row][c4 + 2] = (_Float16)v.z;
        Wh[row][c4 + 3] = (_Float16)v.w;
      }
    }
    __syncthreads();

    v16h a;
#pragma unroll
    for (int i = 0; i < 16; ++i) a[i] = arow[k0 + a_col(i, g)];

#pragma unroll
    for (int nt = 0; nt < 4; ++nt) {
      v16h bf;
#pragma unroll
      for (int i = 0; i < 16; ++i) bf[i] = Wh[nt * 16 + lm][b_row(i, g)];
      acc[nt] = wmma_f16(a, bf, acc[nt]);
    }
    __syncthreads();
  }

#pragma unroll
  for (int nt = 0; nt < 4; ++nt)
#pragma unroll
    for (int r = 0; r < 8; ++r) {
      const int m = m0 + wave * 16 + r + 8 * g;
      const int n = n0 + nt * 16 + lm;
      out[(size_t)m * EMBED + n] = acc[nt][r] + bias[n];
    }
}

// ---------------------------------------------------------------------------
extern "C" void kernel_launch(void* const* d_in, const int* in_sizes, int n_in,
                              void* d_out, int out_size, void* d_ws,
                              size_t ws_size, hipStream_t stream) {
  (void)in_sizes; (void)n_in; (void)out_size; (void)ws_size;
  const float* x     = (const float*)d_in[0];
  const float* Wqkv  = (const float*)d_in[1];
  const float* Wproj = (const float*)d_in[2];
  const float* bproj = (const float*)d_in[3];
  float* out = (float*)d_out;

  _Float16* ws = (_Float16*)d_ws;
  const size_t per = (size_t)BATCH * NHEAD * SEQ * HDIM;  // == B*S*E elements
  _Float16* Qw = ws;
  _Float16* Kw = Qw + per;
  _Float16* Vw = Kw + per;
  _Float16* Ao = Vw + per;   // [B,S,E] f16

  qkv_gemm<<<dim3(M_TOTAL / 64, (3 * EMBED) / 64), 128, 0, stream>>>(
      x, Wqkv, Qw, Kw, Vw);
  flash_attn<<<dim3(BATCH * NHEAD * (SEQ / 64)), 128, 0, stream>>>(
      Qw, Kw, Vw, Ao);
  proj_gemm<<<dim3(M_TOTAL / 64, EMBED / 64), 128, 0, stream>>>(
      Ao, Wproj, bproj, out);
}
